// PatchMerging_3186865733684
// MI455X (gfx1250) — matrix operand order repaired and checked by reading, not compiled, not run on hardware
//
#include <hip/hip_runtime.h>
#include <hip/hip_bf16.h>

typedef __attribute__((ext_vector_type(16))) __bf16 v16bf;
typedef __attribute__((ext_vector_type(8)))  float  v8f;
typedef __attribute__((ext_vector_type(4)))  float  f32x4;   // native vector for NT loads

#define TOK_PER_WG 64
#define LDSA_BYTES (4*24*32*16*2)            /* 4 Mtiles x 24 Ksteps x 32 lanes x 16 bf16 = 98304 B */
#define SMEM_BYTES (LDSA_BYTES + 2*64*4)     /* + mu[64], rstd[64] */
#define WSB_BYTES  (24*12*32*16*2)           /* Wg bf16 swizzled = 294912 B */

__device__ __forceinline__ unsigned bf16rne(float f) {
    unsigned u = __builtin_bit_cast(unsigned, f);
    return (u + 0x7FFFu + ((u >> 16) & 1u)) >> 16;   // round-to-nearest-even bf16
}

// ---- Precompute 1: Wg = ln_w * W, stored bf16 in exact WMMA B-operand (32x16 KxN) swizzle.
// flat index = (((ks*12 + nt)*32 + lane)*16 + e);  lane g=lane>>4 selects K half-block,
// element e = 2r+h -> k = ks*32 + 8*g + (e&7) + 16*(e>>3) ; n = nt*16 + (lane&15)
__global__ void wg_build(const float* __restrict__ W, const float* __restrict__ lnw,
                         unsigned short* __restrict__ wsB) {
    int id = blockIdx.x * 256 + threadIdx.x;          // 576*256 == 147456 exactly
    int e    = id & 15;
    int lane = (id >> 4) & 31;
    int nt   = (id >> 9) % 12;
    int ks   = id / (16 * 32 * 12);
    int r = e >> 1, h = e & 1, g = lane >> 4;
    int n = nt * 16 + (lane & 15);
    int k = ks * 32 + g * 8 + ((2 * r + h) & 7) + 16 * ((2 * r + h) >> 3);
    float v = W[n * 768 + k] * lnw[k];
    wsB[id] = (unsigned short)bf16rne(v);
}

// ---- Precompute 2: s1[n] = sum_k ln_w[k]*W[n,k],  s2[n] = sum_k ln_b[k]*W[n,k]
__global__ void s_build(const float* __restrict__ W, const float* __restrict__ lnw,
                        const float* __restrict__ lnb,
                        float* __restrict__ s1, float* __restrict__ s2) {
    int n = threadIdx.x;                               // 192 threads, 1 block
    float a = 0.f, b = 0.f;
    for (int k = 0; k < 768; ++k) {
        float w = W[n * 768 + k];
        a += w * lnw[k];
        b += w * lnb[k];
    }
    s1[n] = a; s2[n] = b;
}

// ---- Main fused kernel: gather + stats + bf16 A-staging, then WMMA GEMM + LN epilogue.
__global__ __launch_bounds__(256, 2)
void pm_fused_kernel(const float* __restrict__ x,
                     const unsigned short* __restrict__ wsB,
                     const float* __restrict__ s1g, const float* __restrict__ s2g,
                     float* __restrict__ out) {
    extern __shared__ char smem[];
    float* ldsMu = (float*)(smem + LDSA_BYTES);
    float* ldsRs = ldsMu + 64;

    const int tid = threadIdx.x;
    const int t0  = blockIdx.x * TOK_PER_WG;

    // ---------------- Phase 1: gather x, fp32 stats, quantize -> LDS in A swizzle ----------------
    {
        const int tm = tid >> 2;          // local token 0..63 (4 threads per token)
        const int q  = tid & 3;
        const int t  = t0 + tm;
        const int w2 = t & 31, h2 = (t >> 5) & 31, d2 = (t >> 10) & 31, b = t >> 15;
        const int m  = tm & 15, mt = tm >> 4;
        float s = 0.f, ss = 0.f;
        #pragma unroll 1
        for (int p = 0; p < 8; ++p) {     // patch order matches itertools.product((0,1)^3)
            const int p0 = p >> 2, p1 = (p >> 1) & 1, p2 = p & 1;
            const float* seg = x + ((((b * 64 + 2 * d2 + p0) * 64 + 2 * h2 + p1) * 64
                                     + (2 * w2 + p2)) * 96);
            #pragma unroll                // 6 NT float4 loads clause together, then 6 ds stores
            for (int i = 0; i < 6; ++i) {
                const int f = q + 4 * i;          // float4 index within the 96-ch segment
                const int c = 4 * f;
                const f32x4 v = __builtin_nontemporal_load((const f32x4*)(seg + c));
                s  += v.x + v.y + v.z + v.w;
                ss += v.x * v.x + v.y * v.y + v.z * v.z + v.w * v.w;
                // A-operand swizzle (16-bit A 16x32, ISA 7.12.2)
                const int k  = 96 * p + c;
                const int ks = k >> 5, kk = k & 31;
                const int q2 = kk >> 1;                       // even (k % 4 == 0)
                const int g  = (q2 >> 2) & 1;
                const int r  = (q2 & 3) + ((q2 >> 3) << 2);   // even
                const int e  = 2 * r;                         // multiple of 4 -> 8B aligned
                const int byteoff = ((((mt * 24 + ks) * 32) + (m + 16 * g)) * 16 + e) * 2;
                const unsigned lo = bf16rne(v.x) | (bf16rne(v.y) << 16);
                const unsigned hi = bf16rne(v.z) | (bf16rne(v.w) << 16);
                *(uint2*)(smem + byteoff) = make_uint2(lo, hi);
            }
        }
        // reduce the 4 threads of this token (contiguous lanes in a wave32)
        s  += __shfl_xor(s, 1);  s  += __shfl_xor(s, 2);
        ss += __shfl_xor(ss, 1); ss += __shfl_xor(ss, 2);
        if (q == 0) {
            const float mu  = s * (1.f / 768.f);
            const float var = ss * (1.f / 768.f) - mu * mu;
            ldsMu[tm] = mu;
            ldsRs[tm] = rsqrtf(var + 1e-5f);
        }
    }
    __syncthreads();

    // ------------- Phase 2: WMMA GEMM, wave = 32M x 48N strip (2 Mtiles x 3 Ntiles) -------------
    // 8 waves = 2 M-halves x 4 N-quarters; each B tile feeds 2 WMMAs -> halved L2 B traffic.
    const int w    = tid >> 5;
    const int lane = tid & 31;
    const int mh   = w >> 2;           // 0..1 -> M tiles mh*2, mh*2+1
    const int nq   = w & 3;            // 0..3 -> N tiles nq*3 .. nq*3+2

    v8f acc[2][3];
    #pragma unroll
    for (int m = 0; m < 2; ++m)
        #pragma unroll
        for (int i = 0; i < 3; ++i)
            acc[m][i] = (v8f){0.f,0.f,0.f,0.f,0.f,0.f,0.f,0.f};

    float s1v[3], s2v[3];
    #pragma unroll
    for (int i = 0; i < 3; ++i) {
        const int n = (nq * 3 + i) * 16 + (lane & 15);
        s1v[i] = s1g[n];
        s2v[i] = s2g[n];
    }

    #pragma unroll 1
    for (int ks = 0; ks < 24; ++ks) {
        union { uint4 u[2]; v16bf v; } A[2];
        #pragma unroll
        for (int m = 0; m < 2; ++m) {
            const int mt = mh * 2 + m;
            const uint4* ap = (const uint4*)(smem + ((mt * 24 + ks) * 32 + lane) * 32);
            A[m].u[0] = ap[0]; A[m].u[1] = ap[1];       // ds_load_b128 x2 (contiguous)
        }
        #pragma unroll
        for (int i = 0; i < 3; ++i) {
            const int nt = nq * 3 + i;
            union { uint4 u[2]; v16bf v; } Bm;
            const uint4* bp = (const uint4*)wsB + (((ks * 12 + nt) * 32 + lane) * 2);
            Bm.u[0] = bp[0]; Bm.u[1] = bp[1];           // global_load_b128 x2 (L2-resident)
            acc[0][i] = __builtin_amdgcn_wmma_f32_16x16x32_bf16(
                            false, A[0].v, false, Bm.v, (short)0, acc[0][i], false, false);
            acc[1][i] = __builtin_amdgcn_wmma_f32_16x16x32_bf16(
                            false, A[1].v, false, Bm.v, (short)0, acc[1][i], false, false);
        }
    }

    // ---------------- Epilogue: out = rstd*(acc - mu*s1) + s2  (NT stores) ----------------
    const int g  = lane >> 4;
    const int nl = lane & 15;
    #pragma unroll
    for (int m = 0; m < 2; ++m) {
        const int mt = mh * 2 + m;
        #pragma unroll
        for (int j = 0; j < 8; ++j) {
            const int tmj = mt * 16 + j + 8 * g;        // C layout: VGPR j -> M=j / M=8+j
            const float mu = ldsMu[tmj];
            const float rs = ldsRs[tmj];
            const size_t row = (size_t)(t0 + tmj) * 192;
            #pragma unroll
            for (int i = 0; i < 3; ++i) {
                const int n = (nq * 3 + i) * 16 + nl;
                const float val = rs * (acc[m][i][j] - mu * s1v[i]) + s2v[i];
                __builtin_nontemporal_store(val, &out[row + n]);
            }
        }
    }
}

extern "C" void kernel_launch(void* const* d_in, const int* in_sizes, int n_in,
                              void* d_out, int out_size, void* d_ws, size_t ws_size,
                              hipStream_t stream) {
    const float* x   = (const float*)d_in[0];
    const float* lnw = (const float*)d_in[1];
    const float* lnb = (const float*)d_in[2];
    const float* W   = (const float*)d_in[3];
    float* out = (float*)d_out;

    unsigned short* wsB = (unsigned short*)d_ws;
    float* s1 = (float*)((char*)d_ws + WSB_BYTES);
    float* s2 = s1 + 192;

    (void)in_sizes; (void)n_in; (void)out_size; (void)ws_size;

    static bool attr_set = false;   // host-side only; not a stream op, capture-safe
    if (!attr_set) {
        (void)hipFuncSetAttribute((const void*)pm_fused_kernel,
                                  hipFuncAttributeMaxDynamicSharedMemorySize, SMEM_BYTES);
        attr_set = true;
    }

    wg_build<<<576, 256, 0, stream>>>(W, lnw, wsB);
    s_build<<<1, 192, 0, stream>>>(W, lnw, lnb, s1, s2);
    pm_fused_kernel<<<1024, 256, SMEM_BYTES, stream>>>(x, wsB, s1, s2, out);
}
